// Attention_16441134809204
// MI455X (gfx1250) — matrix operand order repaired and checked by reading, not compiled, not run on hardware
//
#include <hip/hip_runtime.h>
#include <hip/hip_bf16.h>

// ---------------------------------------------------------------------------
// Attention layer for MI455X (gfx1250, wave32, WMMA f32_16x16x32_f16)
// Pipeline: cvt -> QKV GEMM (WMMA) -> RoPE/pack -> flash-attn (WMMA) -> out GEMM
// ---------------------------------------------------------------------------

typedef __attribute__((ext_vector_type(16))) _Float16 v16h;
typedef __attribute__((ext_vector_type(8)))  _Float16 v8h;
typedef __attribute__((ext_vector_type(8)))  float    v8f;

#define WMMA_F16(a, b, c) \
  __builtin_amdgcn_wmma_f32_16x16x32_f16(false, (a), false, (b), (short)0, (c), false, false)

// A fragment (16x32, MxK): lane holds row M=lane%16; elements 0..7 at K=8*half,
// elements 8..15 at K=16+8*half (ISA 7.12.2, 16-bit A-matrix table).
__device__ inline v16h load_a_frag(const _Float16* base, int ld) {
  const int lane = threadIdx.x & 31;
  const int row  = lane & 15;
  const int half = lane >> 4;
  const _Float16* p = base + (size_t)row * ld + 8 * half;
  v8h lo = *(const v8h*)(p);
  v8h hi = *(const v8h*)(p + 16);
  v16h r;
#pragma unroll
  for (int i = 0; i < 8; ++i) { r[i] = lo[i]; r[i + 8] = hi[i]; }
  return r;
}

// B fragment (32x16, KxN), sourced column-major ([N][K] buffer): lane holds
// col N=lane%16; elements 0..7 at K=8*half, elements 8..15 at K=16+8*half.
__device__ inline v16h load_b_frag(const _Float16* base, int ld) {
  const int lane = threadIdx.x & 31;
  const int col  = lane & 15;
  const int half = lane >> 4;
  const _Float16* p = base + (size_t)col * ld + 8 * half;
  v8h lo = *(const v8h*)(p);
  v8h hi = *(const v8h*)(p + 16);
  v16h r;
#pragma unroll
  for (int i = 0; i < 8; ++i) { r[i] = lo[i]; r[i + 8] = hi[i]; }
  return r;
}

// ---------------------------------------------------------------------------
// Elementwise conversion kernels
// ---------------------------------------------------------------------------
__global__ void cvt_f32_f16(const float* __restrict__ in, _Float16* __restrict__ out, int n) {
  int i = blockIdx.x * 256 + threadIdx.x;
  if (i < n) out[i] = (_Float16)in[i];
}

// in: [K][N] f32 row-major  ->  out: [N][K] f16 (transposed)
__global__ void cvt_transpose(const float* __restrict__ in, _Float16* __restrict__ out,
                              int K, int N) {
  int i = blockIdx.x * 256 + threadIdx.x;
  if (i < K * N) {
    int n = i / K;
    int k = i - n * K;
    out[i] = (_Float16)in[(size_t)k * N + n];
  }
}

// ---------------------------------------------------------------------------
// GEMM: C[M][N] f32 = A[M][K] f16  x  Bt[N][K] f16 (B stored transposed)
// Block tile 128x128, 8 waves (256 thr), wave tile 64x32, K-step 32,
// double-buffered LDS: one barrier per K-step, next tile loads overlap WMMAs.
// ---------------------------------------------------------------------------
__global__ void __launch_bounds__(256) gemm_f16_f32(
    const _Float16* __restrict__ A, const _Float16* __restrict__ Bt,
    float* __restrict__ C, int M, int N, int K) {
  __shared__ __align__(16) _Float16 sA[2][128][32];
  __shared__ __align__(16) _Float16 sB[2][128][32];

  const int t  = threadIdx.x;
  const int m0 = blockIdx.x * 128;
  const int n0 = blockIdx.y * 128;
  const int w  = t >> 5;
  const int wm = (w >> 2) * 64;     // 0,64
  const int wn = (w & 3) * 32;      // 0,32,64,96

  const int row = t >> 1, kk = (t & 1) * 16;   // 128 rows x 32 k, 32 f16/thread

  const _Float16* ga = A  + (size_t)(m0 + row) * K + kk;
  const _Float16* gb = Bt + (size_t)(n0 + row) * K + kk;

  // prologue: tile 0 -> buffer 0
  *(v8h*)&sA[0][row][kk]     = *(const v8h*)(ga);
  *(v8h*)&sA[0][row][kk + 8] = *(const v8h*)(ga + 8);
  *(v8h*)&sB[0][row][kk]     = *(const v8h*)(gb);
  *(v8h*)&sB[0][row][kk + 8] = *(const v8h*)(gb + 8);

  v8f acc[4][2];
#pragma unroll
  for (int i = 0; i < 4; ++i) { acc[i][0] = (v8f){}; acc[i][1] = (v8f){}; }

  int cur = 0;
  for (int kb = 0; kb < K; kb += 32) {
    __syncthreads();  // buffer[cur] fully written
    if (kb + 32 < K) {
      const _Float16* na = ga + kb + 32;
      const _Float16* nb = gb + kb + 32;
      *(v8h*)&sA[cur ^ 1][row][kk]     = *(const v8h*)(na);
      *(v8h*)&sA[cur ^ 1][row][kk + 8] = *(const v8h*)(na + 8);
      *(v8h*)&sB[cur ^ 1][row][kk]     = *(const v8h*)(nb);
      *(v8h*)&sB[cur ^ 1][row][kk + 8] = *(const v8h*)(nb + 8);
    }
    v16h b0 = load_b_frag(&sB[cur][wn][0], 32);
    v16h b1 = load_b_frag(&sB[cur][wn + 16][0], 32);
#pragma unroll
    for (int i = 0; i < 4; ++i) {
      v16h a = load_a_frag(&sA[cur][wm + 16 * i][0], 32);
      acc[i][0] = WMMA_F16(a, b0, acc[i][0]);
      acc[i][1] = WMMA_F16(a, b1, acc[i][1]);
    }
    cur ^= 1;
  }

  const int lane = t & 31, col = lane & 15, half = lane >> 4;
#pragma unroll
  for (int i = 0; i < 4; ++i) {
#pragma unroll
    for (int r = 0; r < 8; ++r) {
      int m = m0 + wm + 16 * i + r + 8 * half;   // D: vgpr r <-> M = r + 8*half
      C[(size_t)m * N + n0 + wn + col]      = acc[i][0][r];
      C[(size_t)m * N + n0 + wn + 16 + col] = acc[i][1][r];
    }
  }
}

// ---------------------------------------------------------------------------
// Pack fp32 KV cache into f16: Kh[bh][0..1023][d] row-major, Vt[bh][d][0..1023].
// ---------------------------------------------------------------------------
__global__ void pack_cache(const float* __restrict__ ck, const float* __restrict__ cv,
                           _Float16* __restrict__ Kh, _Float16* __restrict__ Vt) {
  int i = blockIdx.x * 256 + threadIdx.x;
  int d  = i & 127;
  int s  = (i >> 7) & 1023;
  int bh = i >> 17;
  Kh[((size_t)bh * 2048 + s) * 128 + d] = (_Float16)ck[i];
  Vt[((size_t)bh * 128 + d) * 2048 + s] = (_Float16)cv[i];
}

// ---------------------------------------------------------------------------
// RoPE + pack new Q/K/V. qkv: [4096][6144] f32, per (b,s): head h at col h*384,
// [0:128)=q [128:256)=k [256:384)=v.  offset = 1024.
// ---------------------------------------------------------------------------
__global__ void rope_pack(const float* __restrict__ qkv, _Float16* __restrict__ Qh,
                          _Float16* __restrict__ Kh, _Float16* __restrict__ Vt) {
  int i = blockIdx.x * 256 + threadIdx.x;
  int d = i & 63;
  int s = (i >> 6) & 1023;
  int h = (i >> 16) & 15;
  int b = i >> 20;

  const float* row = qkv + (size_t)(b * 1024 + s) * 6144 + h * 384;
  float q1 = row[d],       q2 = row[d + 64];
  float k1 = row[128 + d], k2 = row[192 + d];
  float v1 = row[256 + d], v2 = row[320 + d];

  float invf = __powf(10000.0f, -(float)d * (1.0f / 64.0f));
  float ang  = (float)(1024 + s) * invf;
  float c = cosf(ang), sn = sinf(ang);

  int bh = b * 16 + h;
  _Float16* qo = Qh + ((size_t)bh * 1024 + s) * 128;
  qo[d]      = (_Float16)(q1 * c - q2 * sn);
  qo[d + 64] = (_Float16)(q2 * c + q1 * sn);

  _Float16* ko = Kh + ((size_t)bh * 2048 + 1024 + s) * 128;
  ko[d]      = (_Float16)(k1 * c - k2 * sn);
  ko[d + 64] = (_Float16)(k2 * c + k1 * sn);

  _Float16* vo = Vt + (size_t)bh * 128 * 2048;
  vo[(size_t)d        * 2048 + 1024 + s] = (_Float16)v1;
  vo[(size_t)(d + 64) * 2048 + 1024 + s] = (_Float16)v2;
}

// ---------------------------------------------------------------------------
// Flash attention: 1 wave per (b, h, 16-query tile). 16 WMMAs per 32-key block.
// Single-wave workgroup => barriers are S_NOP (ISA 3.1).
// Qh[bh][1024][128], Kh[bh][2048][128], Vt[bh][128][2048] -> Oh[b*1024+s][h*128+d]
// ---------------------------------------------------------------------------
__global__ void __launch_bounds__(32) attn_kernel(
    const _Float16* __restrict__ Qh, const _Float16* __restrict__ Kh,
    const _Float16* __restrict__ Vt, _Float16* __restrict__ Oh) {
  __shared__ __align__(16) _Float16 Plds[16][32];

  const int idx = blockIdx.x;
  const int qt  = idx & 63;
  const int bh  = idx >> 6;
  const int h   = bh & 15;
  const int b   = bh >> 4;
  const int lane = threadIdx.x & 31;
  const int col  = lane & 15, half = lane >> 4;

  const _Float16* qbase = Qh + ((size_t)bh * 1024 + qt * 16) * 128;
  const _Float16* kbase = Kh + (size_t)bh * 2048 * 128;
  const _Float16* vbase = Vt + (size_t)bh * 128 * 2048;

  v16h aq[4];
#pragma unroll
  for (int i = 0; i < 4; ++i) aq[i] = load_a_frag(qbase + 32 * i, 128);

  float mrow[8], lrow[8];
  v8f acc[8];
#pragma unroll
  for (int r = 0; r < 8; ++r) { mrow[r] = -1e30f; lrow[r] = 0.0f; }
#pragma unroll
  for (int j = 0; j < 8; ++j) acc[j] = (v8f){};

  const float scale = 0.08838834764831845f;  // 1/sqrt(128)

  for (int kb = 0; kb < 2048; kb += 32) {
    // scores: 16 queries x 32 keys, contraction over d=128
    v8f s0 = {}, s1 = {};
#pragma unroll
    for (int i = 0; i < 4; ++i) {
      v16h bk0 = load_b_frag(kbase + (size_t)kb * 128 + 32 * i, 128);
      v16h bk1 = load_b_frag(kbase + (size_t)(kb + 16) * 128 + 32 * i, 128);
      s0 = WMMA_F16(aq[i], bk0, s0);
      s1 = WMMA_F16(aq[i], bk1, s1);
    }

    // online softmax; row m = r + 8*half lives across a 16-lane half-group
#pragma unroll
    for (int r = 0; r < 8; ++r) {
      float x0 = s0[r] * scale, x1 = s1[r] * scale;
      float mb = fmaxf(x0, x1);
#pragma unroll
      for (int off = 1; off < 16; off <<= 1) mb = fmaxf(mb, __shfl_xor(mb, off, 32));
      float mn   = fmaxf(mrow[r], mb);
      float corr = __expf(mrow[r] - mn);
      float p0 = __expf(x0 - mn), p1 = __expf(x1 - mn);
      float rs = p0 + p1;
#pragma unroll
      for (int off = 1; off < 16; off <<= 1) rs += __shfl_xor(rs, off, 32);
      lrow[r] = lrow[r] * corr + rs;
      mrow[r] = mn;
#pragma unroll
      for (int j = 0; j < 8; ++j) acc[j][r] = acc[j][r] * corr;
      Plds[r + 8 * half][col]      = (_Float16)p0;
      Plds[r + 8 * half][16 + col] = (_Float16)p1;
    }
    __syncthreads();

    // P (16x32 over keys) as A fragment; V fragments direct from transposed global
    v16h ap = load_a_frag(&Plds[0][0], 32);
#pragma unroll
    for (int j = 0; j < 8; ++j) {
      v16h bv = load_b_frag(vbase + (size_t)(j * 16) * 2048 + kb, 2048);
      acc[j] = WMMA_F16(ap, bv, acc[j]);
    }
    __syncthreads();
  }

  // normalize and write Oh[b*1024+s][h*128+d]
#pragma unroll
  for (int r = 0; r < 8; ++r) {
    float inv = 1.0f / lrow[r];
    int s = qt * 16 + r + 8 * half;
    size_t orow = ((size_t)b * 1024 + s) * 2048 + h * 128;
#pragma unroll
    for (int j = 0; j < 8; ++j)
      Oh[orow + j * 16 + col] = (_Float16)(acc[j][r] * inv);
  }
}

// ---------------------------------------------------------------------------
// Host launcher
// ---------------------------------------------------------------------------
extern "C" void kernel_launch(void* const* d_in, const int* in_sizes, int n_in,
                              void* d_out, int out_size, void* d_ws, size_t ws_size,
                              hipStream_t stream) {
  (void)in_sizes; (void)n_in; (void)out_size; (void)ws_size;
  const float* x       = (const float*)d_in[0];
  const float* cache_k = (const float*)d_in[1];
  const float* cache_v = (const float*)d_in[2];
  const float* w_qkv   = (const float*)d_in[3];
  const float* w_o     = (const float*)d_in[4];
  float* out = (float*)d_out;

  char* ws = (char*)d_ws;
  size_t off = 0;
  auto alloc = [&](size_t bytes) -> void* {
    void* p = ws + off;
    off += (bytes + 255) & ~(size_t)255;
    return p;
  };

  _Float16* Xh    = (_Float16*)alloc(4096ull * 2048 * 2);     // 16 MiB
  _Float16* WqkvT = (_Float16*)alloc(6144ull * 2048 * 2);     // 24 MiB
  _Float16* WoT   = (_Float16*)alloc(2048ull * 2048 * 2);     //  8 MiB
  float*    QKV   = (float*)   alloc(4096ull * 6144 * 4);     // 96 MiB
  _Float16* Qh    = (_Float16*)alloc(64ull * 1024 * 128 * 2); // 16 MiB
  _Float16* Kh    = (_Float16*)alloc(64ull * 2048 * 128 * 2); // 32 MiB
  _Float16* Vt    = (_Float16*)alloc(64ull * 128 * 2048 * 2); // 32 MiB
  _Float16* Oh    = Xh;  // X no longer needed after the QKV GEMM

  cvt_f32_f16<<<(4096 * 2048) / 256, 256, 0, stream>>>(x, Xh, 4096 * 2048);
  cvt_transpose<<<(2048 * 6144) / 256, 256, 0, stream>>>(w_qkv, WqkvT, 2048, 6144);
  cvt_transpose<<<(2048 * 2048) / 256, 256, 0, stream>>>(w_o, WoT, 2048, 2048);

  // qkv = x @ w_qkv : M=4096, N=6144, K=2048
  gemm_f16_f32<<<dim3(4096 / 128, 6144 / 128), 256, 0, stream>>>(
      Xh, WqkvT, QKV, 4096, 6144, 2048);

  pack_cache<<<(64 * 1024 * 128) / 256, 256, 0, stream>>>(cache_k, cache_v, Kh, Vt);
  rope_pack<<<(4 * 16 * 1024 * 64) / 256, 256, 0, stream>>>(QKV, Qh, Kh, Vt);

  attn_kernel<<<4 * 16 * 64, 32, 0, stream>>>(Qh, Kh, Vt, Oh);

  // out = attn_out @ w_o : M=4096, N=2048, K=2048
  gemm_f16_f32<<<dim3(4096 / 128, 2048 / 128), 256, 0, stream>>>(
      Oh, WoT, out, 4096, 2048, 2048);
}
